// VectorQuantizer_2095944040838
// MI455X (gfx1250) — matrix operand order repaired and checked by reading, not compiled, not run on hardware
//
#include <hip/hip_runtime.h>
#include <hip/hip_bf16.h>

typedef __attribute__((ext_vector_type(16))) _Float16 v16h;
typedef __attribute__((ext_vector_type(8)))  float    v8f;

#define VQ_K    512      // codebook entries
#define VQ_D    64       // embedding dim
#define CB_STRIDE 72     // padded row stride in halves (36 dwords -> conflict-free ds_load_b128)
#define ROWS_PER_BLOCK 128  // 8 waves * 16 rows

// LDS layout (dynamic): padded f16 codebook, then 512 f32 norms, then 8 f32 wave sums
#define LDS_CB_BYTES  (VQ_K * CB_STRIDE * 2)   // 73728
#define LDS_CN_BYTES  (VQ_K * 4)               // 2048
#define LDS_WS_BYTES  (8 * 4)
#define LDS_TOTAL     (LDS_CB_BYTES + LDS_CN_BYTES + LDS_WS_BYTES)

__global__ __launch_bounds__(256) void vq_main_kernel(
    const float* __restrict__ x,       // [N, 64]
    const float* __restrict__ cb,      // [512, 64]
    float* __restrict__ out,           // [N, 64]
    float* __restrict__ partial)       // [gridDim.x]
{
    extern __shared__ char smem[];
    _Float16* s_cb = (_Float16*)smem;                       // [512][72] f16 (padded)
    float*    s_cn = (float*)(smem + LDS_CB_BYTES);         // [512] norms
    float*    s_ws = (float*)(smem + LDS_CB_BYTES + LDS_CN_BYTES); // [8]

    const int tid  = threadIdx.x;
    const int lane = tid & 31;
    const int wave = tid >> 5;
    const int half = lane >> 4;   // 0: lanes 0-15, 1: lanes 16-31
    const int l15  = lane & 15;

    // ---- Stage codebook as f16 into LDS (padded rows), exact f32 norms ----
    {
        const float2* cb2 = (const float2*)cb;   // 16384 float2
        for (int i = 0; i < 64; ++i) {
            int j = i * 256 + tid;               // interleaved for coalescing
            float2 v = cb2[j];
            int e = j >> 5;                      // entry  (32 float2 per row)
            int d = (j & 31) * 2;                // dim
            s_cb[e * CB_STRIDE + d]     = (_Float16)v.x;
            s_cb[e * CB_STRIDE + d + 1] = (_Float16)v.y;
        }
        for (int e = tid; e < VQ_K; e += 256) {
            const float* row = cb + e * VQ_D;
            float s = 0.f;
            #pragma unroll
            for (int d = 0; d < VQ_D; ++d) s += row[d] * row[d];
            s_cn[e] = s;
        }
    }
    __syncthreads();

    // ---- Load A fragment: rows of (-2*x), f16, 16x64 per wave ----
    // A layout (16-bit A 16x32): lane M = lane&15; VGPR v<4 -> K = 2v + 8*half (+0,1)
    //                                         v>=4 -> K = 16 + 2(v-4) + 8*half (+0,1)
    const int rowbase = (int)blockIdx.x * ROWS_PER_BLOCK + wave * 16;
    const float* xrow = x + (size_t)(rowbase + l15) * VQ_D;

    v16h a_lo, a_hi;
    #pragma unroll
    for (int v = 0; v < 8; ++v) {
        int kb = ((v < 4) ? (2 * v) : (16 + 2 * (v - 4))) + 8 * half;
        float2 x0 = *(const float2*)(xrow + kb);        // dims [kb, kb+1]
        float2 x1 = *(const float2*)(xrow + 32 + kb);   // dims [32+kb, 32+kb+1]
        a_lo[2 * v]     = (_Float16)(-2.0f * x0.x);
        a_lo[2 * v + 1] = (_Float16)(-2.0f * x0.y);
        a_hi[2 * v]     = (_Float16)(-2.0f * x1.x);
        a_hi[2 * v + 1] = (_Float16)(-2.0f * x1.y);
    }

    // ---- Scan all 512 codes, two 16-column tiles per iteration ----
    float best[8];
    int   bidx[8];
    #pragma unroll
    for (int v = 0; v < 8; ++v) { best[v] = 3.4e38f; bidx[v] = 0; }

    for (int kt = 0; kt < VQ_K / 16; kt += 2) {
        const int n0 = kt * 16 + l15;
        const int n1 = n0 + 16;
        const float cn0 = s_cn[n0];
        const float cn1 = s_cn[n1];

        v8f c0, c1;
        #pragma unroll
        for (int v = 0; v < 8; ++v) { c0[v] = cn0; c1[v] = cn1; }

        // B layout (16-bit B 32x16): lane N = lane&15; VGPR v<4 -> K = 2v + 8*half (+0,1), v>=4 -> K = 16 + 2(v-4) + 8*half (+0,1)
        v16h b0_lo, b0_hi, b1_lo, b1_hi;
        const _Float16* crow0 = s_cb + n0 * CB_STRIDE;
        const _Float16* crow1 = s_cb + n1 * CB_STRIDE;
        #pragma unroll
        for (int v = 0; v < 8; ++v) {
            int kb = ((v < 4) ? (2 * v) : (16 + 2 * (v - 4))) + 8 * half;
            b0_lo[2 * v]     = crow0[kb];
            b0_lo[2 * v + 1] = crow0[kb + 1];
            b0_hi[2 * v]     = crow0[32 + kb];
            b0_hi[2 * v + 1] = crow0[32 + kb + 1];
            b1_lo[2 * v]     = crow1[kb];
            b1_lo[2 * v + 1] = crow1[kb + 1];
            b1_hi[2 * v]     = crow1[32 + kb];
            b1_hi[2 * v + 1] = crow1[32 + kb + 1];
        }

        // D = ||c||^2 + (-2x)·c over K=64; two independent WMMA chains overlap
        c0 = __builtin_amdgcn_wmma_f32_16x16x32_f16(false, a_lo, false, b0_lo,
                                                    (short)0, c0, false, false);
        c1 = __builtin_amdgcn_wmma_f32_16x16x32_f16(false, a_lo, false, b1_lo,
                                                    (short)0, c1, false, false);
        c0 = __builtin_amdgcn_wmma_f32_16x16x32_f16(false, a_hi, false, b0_hi,
                                                    (short)0, c0, false, false);
        c1 = __builtin_amdgcn_wmma_f32_16x16x32_f16(false, a_hi, false, b1_hi,
                                                    (short)0, c1, false, false);

        // branchless running min (strict '<' keeps earliest index, n ascending)
        #pragma unroll
        for (int v = 0; v < 8; ++v) {
            bool t0 = c0[v] < best[v];
            best[v] = t0 ? c0[v] : best[v];
            bidx[v] = t0 ? n0    : bidx[v];
            bool t1 = c1[v] < best[v];
            best[v] = t1 ? c1[v] : best[v];
            bidx[v] = t1 ? n1    : bidx[v];
        }
    }

    // ---- Branchless cross-lane min+argmin within each 16-lane half ----
    #pragma unroll
    for (int m = 1; m < 16; m <<= 1) {
        #pragma unroll
        for (int v = 0; v < 8; ++v) {
            float ov = __shfl_xor(best[v], m, 32);
            int   oi = __shfl_xor(bidx[v], m, 32);
            int take = (int)(ov < best[v]) | ((int)(ov == best[v]) & (int)(oi < bidx[v]));
            best[v] = take ? ov : best[v];
            bidx[v] = take ? oi : bidx[v];
        }
    }
    // lane 0 holds argmin for rows M=0..7 (v), lane 16 for rows M=8..15 (v)

    // ---- Gather exact f32 codebook rows, write output, accumulate loss ----
    float acc = 0.f;
    #pragma unroll
    for (int r = 0; r < 16; ++r) {
        const int idx = __builtin_amdgcn_readlane(bidx[r & 7], (r & 8) ? 16 : 0); // SGPR
        const size_t orow = (size_t)(rowbase + r) * VQ_D;
        float2 q  = ((const float2*)(cb + (size_t)idx * VQ_D))[lane]; // 32 lanes x2 = 64
        float2 xv = ((const float2*)(x + orow))[lane];
        ((float2*)(out + orow))[lane] = q;
        float d0 = q.x - xv.x, d1 = q.y - xv.y;
        acc += d0 * d0 + d1 * d1;
    }

    // deterministic wave sum, then block sum -> partial[block]
    #pragma unroll
    for (int m = 1; m < 32; m <<= 1) acc += __shfl_xor(acc, m, 32);
    if (lane == 0) s_ws[wave] = acc;
    __syncthreads();
    if (tid == 0) {
        float t = 0.f;
        #pragma unroll
        for (int w = 0; w < 8; ++w) t += s_ws[w];
        partial[blockIdx.x] = t;
    }
}

__global__ __launch_bounds__(256) void vq_loss_kernel(
    const float* __restrict__ partial, int nblk, float inv_count,
    float* __restrict__ loss_out)
{
    __shared__ float s[256];
    const int t = threadIdx.x;
    float a = 0.f;
    for (int i = t; i < nblk; i += 256) a += partial[i];
    s[t] = a;
    __syncthreads();
    #pragma unroll
    for (int stp = 128; stp > 0; stp >>= 1) {
        if (t < stp) s[t] += s[t + stp];
        __syncthreads();
    }
    if (t == 0) *loss_out = 1.25f * s[0] * inv_count;  // q_loss + 0.25*e_loss
}

extern "C" void kernel_launch(void* const* d_in, const int* in_sizes, int n_in,
                              void* d_out, int out_size, void* d_ws, size_t ws_size,
                              hipStream_t stream) {
    const float* x  = (const float*)d_in[0];   // [32,64,64,64] f32
    const float* cb = (const float*)d_in[1];   // [512,64] f32
    float* out      = (float*)d_out;           // 8388608 quantized + 1 loss
    float* partial  = (float*)d_ws;

    const int n_elems = in_sizes[0];           // 8388608
    const int rows    = n_elems / VQ_D;        // 131072
    const int nblk    = rows / ROWS_PER_BLOCK; // 1024

    vq_main_kernel<<<nblk, 256, LDS_TOTAL, stream>>>(x, cb, out, partial);

    const float inv_count = 1.0f / (float)n_elems;
    vq_loss_kernel<<<1, 256, 0, stream>>>(partial, nblk, inv_count,
                                          out + (out_size - 1));
}
